// CausalSelfAttention_77429670413042
// MI455X (gfx1250) — matrix operand order repaired and checked by reading, not compiled, not run on hardware
//
#include <hip/hip_runtime.h>

// ---------------------------------------------------------------------------
// Types
// ---------------------------------------------------------------------------
typedef __bf16 bf16_t;
typedef __attribute__((ext_vector_type(8)))  bf16_t         v8bf;
typedef __attribute__((ext_vector_type(16))) bf16_t         v16bf;
typedef __attribute__((ext_vector_type(8)))  float          v8f;
typedef __attribute__((ext_vector_type(8)))  unsigned short v8u;

#define D_MODEL 1024
#define SEQ     4096
#define NHEAD   16
#define HDIM    64

// ---------------------------------------------------------------------------
// Async global->LDS copy (CDNA5 GLOBAL_LOAD_ASYNC_TO_LDS_B128, ASYNCcnt)
// ---------------------------------------------------------------------------
#if defined(__has_builtin)
#  if __has_builtin(__builtin_amdgcn_global_load_async_to_lds_b128)
#    define HAVE_ASYNC_LDS 1
#  endif
#endif
#ifndef HAVE_ASYNC_LDS
#  define HAVE_ASYNC_LDS 0
#endif

#if HAVE_ASYNC_LDS
typedef int gcc_v4i __attribute__((vector_size(4 * sizeof(int))));
typedef __attribute__((address_space(1))) gcc_v4i* global_v4i_p;
typedef __attribute__((address_space(3))) gcc_v4i* lds_v4i_p;
#endif

__device__ __forceinline__ void copy16_g2l(const unsigned short* g,
                                           unsigned short* l) {
#if HAVE_ASYNC_LDS
  __builtin_amdgcn_global_load_async_to_lds_b128(
      (global_v4i_p)(void*)g, (lds_v4i_p)(void*)l, 0, 0);
#else
  *(v8u*)l = *(const v8u*)g;
#endif
}
__device__ __forceinline__ void async_fence() {
#if HAVE_ASYNC_LDS
  asm volatile("s_wait_asynccnt 0x0" ::: "memory");
#endif
}

// ---------------------------------------------------------------------------
// Helpers
// ---------------------------------------------------------------------------
__device__ __forceinline__ unsigned short f2bf(float f) {
  unsigned int u = __float_as_uint(f);
  u += 0x7FFFu + ((u >> 16) & 1u);     // round-to-nearest-even
  return (unsigned short)(u >> 16);
}

// A-matrix (16xK) bf16 fragment: lanes<16 hold K={b..b+7, b+16..b+23}, b=0;
// lanes>=16: b=8.  p points at K=0 of this lane's row; hiA = (lane>=16)?8:0.
__device__ __forceinline__ v16bf load_fragA(const unsigned short* p, int hiA) {
  union { v16bf v; v8bf h[2]; } u;
  u.h[0] = *(const v8bf*)(p + hiA);
  u.h[1] = *(const v8bf*)(p + hiA + 16);
  return u.v;
}
// B-matrix (Kx16) bf16 fragment: lanes<16 hold K={b..b+7, b+16..b+23}, b=0;
// lanes>=16: b=8. p points at K=0 of this lane's column (K-contiguous); hiB = (lane>=16)?8:0.
__device__ __forceinline__ v16bf load_fragB(const unsigned short* p, int hiB) {
  union { v16bf v; v8bf h[2]; } u;
  u.h[0] = *(const v8bf*)(p + hiB);
  u.h[1] = *(const v8bf*)(p + hiB + 16);
  return u.v;
}

__device__ __forceinline__ v8f wmma_bf16(v16bf a, v16bf b, v8f c) {
  return __builtin_amdgcn_wmma_f32_16x16x32_bf16(false, a, false, b,
                                                 (short)0, c, false, false);
}

__device__ __forceinline__ float hmax16(float v) {
#pragma unroll
  for (int m = 1; m < 16; m <<= 1) v = fmaxf(v, __shfl_xor(v, m, 32));
  return v;
}
__device__ __forceinline__ float hsum16(float v) {
#pragma unroll
  for (int m = 1; m < 16; m <<= 1) v += __shfl_xor(v, m, 32);
  return v;
}

// fp32 register prefetch of 8 elements + bf16 commit into LDS
struct A8 { float4 a, b; };
__device__ __forceinline__ void prefA(const float* g, A8& r) {
  r.a = *(const float4*)g;
  r.b = *(const float4*)(g + 4);
}
__device__ __forceinline__ void commitA(const A8& r, unsigned short* s) {
  v8u t;
  t[0] = f2bf(r.a.x); t[1] = f2bf(r.a.y); t[2] = f2bf(r.a.z); t[3] = f2bf(r.a.w);
  t[4] = f2bf(r.b.x); t[5] = f2bf(r.b.y); t[6] = f2bf(r.b.z); t[7] = f2bf(r.b.w);
  *(v8u*)s = t;
}

// ---------------------------------------------------------------------------
// fp32 [K][N] -> bf16 [N][K] transpose+convert (LDS tiled, 32x32)
// ---------------------------------------------------------------------------
__global__ __launch_bounds__(256) void transpose_cvt_kernel(
    const float* __restrict__ W, unsigned short* __restrict__ Wt, int K, int N) {
  __shared__ unsigned short tile[32][33];
  int kb = blockIdx.y * 32, nb = blockIdx.x * 32;
  int tx = threadIdx.x, ty = threadIdx.y;      // block (32,8)
#pragma unroll
  for (int i = ty; i < 32; i += 8)
    tile[i][tx] = f2bf(W[(size_t)(kb + i) * N + nb + tx]);
  __syncthreads();
#pragma unroll
  for (int i = ty; i < 32; i += 8)
    Wt[(size_t)(nb + i) * K + kb + tx] = tile[tx][i];
}

// ---------------------------------------------------------------------------
// GEMM:  C[M,N] = A[M,K] * Bt[N,K]^T + bias[N]
//   AT = float (convert on stage) or ushort (bf16, async-staged)
//   OUT_BF16: write bf16 (with qscale on cols < qcols) else fp32
// Block: 256 threads (8 waves), tile 128x128, K-step 32, double-buffered LDS.
// ---------------------------------------------------------------------------
template <typename AT, bool OUT_BF16>
__global__ __launch_bounds__(256) void gemm_bias_wmma(
    const AT* __restrict__ A, const unsigned short* __restrict__ Bt,
    const float* __restrict__ bias, void* __restrict__ Cout,
    int M, int N, int Kdim, float qscale, int qcols) {
  constexpr bool A_F32 = (sizeof(AT) == 4);
  __shared__ unsigned short As[2][128 * 32];
  __shared__ unsigned short Bs[2][128 * 32];

  const int tid = threadIdx.x;
  const int lane = tid & 31, wid = tid >> 5;
  const int wm = wid & 3;        // 4 wave-rows of 32
  const int wn = wid >> 2;       // 2 wave-cols of 64
  const int l16 = lane & 15;
  const int hiA = (lane >= 16) ? 8 : 0;
  const int hiB = (lane >= 16) ? 8 : 0;
  const int bm = blockIdx.y * 128, bn = blockIdx.x * 128;

  // each thread stages rows r0 and r0+64 at column chunk kc0 (8 elems)
  const int r0 = tid >> 2;
  const int kc0 = (tid & 3) * 8;
  const AT* Ag0 = A + (size_t)(bm + r0) * Kdim + kc0;
  const AT* Ag1 = A + (size_t)(bm + r0 + 64) * Kdim + kc0;
  const unsigned short* Bg0 = Bt + (size_t)(bn + r0) * Kdim + kc0;
  const unsigned short* Bg1 = Bt + (size_t)(bn + r0 + 64) * Kdim + kc0;
  const int sA0 = r0 * 32 + kc0, sA1 = (r0 + 64) * 32 + kc0;

  v8f acc[2][4];
#pragma unroll
  for (int i = 0; i < 2; ++i)
#pragma unroll
    for (int j = 0; j < 4; ++j) acc[i][j] = (v8f){};

  const int nk = Kdim / 32;

  // ---- prologue: stage tile 0 into buffer 0
  copy16_g2l(Bg0, &Bs[0][sA0]);
  copy16_g2l(Bg1, &Bs[0][sA1]);
  if constexpr (A_F32) {
    A8 p0, p1;
    prefA((const float*)Ag0, p0);
    prefA((const float*)Ag1, p1);
    commitA(p0, &As[0][sA0]);
    commitA(p1, &As[0][sA1]);
  } else {
    copy16_g2l((const unsigned short*)Ag0, &As[0][sA0]);
    copy16_g2l((const unsigned short*)Ag1, &As[0][sA1]);
  }
  async_fence();
  __syncthreads();

  int cur = 0;
  for (int kt = 0; kt < nk; ++kt) {
    const int nxt = cur ^ 1;
    const bool have_next = (kt + 1 < nk);
    A8 p0, p1;
    if (have_next) {
      const int kn = (kt + 1) * 32;
      copy16_g2l(Bg0 + kn, &Bs[nxt][sA0]);
      copy16_g2l(Bg1 + kn, &Bs[nxt][sA1]);
      if constexpr (A_F32) {
        prefA((const float*)Ag0 + kn, p0);
        prefA((const float*)Ag1 + kn, p1);
      } else {
        copy16_g2l((const unsigned short*)Ag0 + kn, &As[nxt][sA0]);
        copy16_g2l((const unsigned short*)Ag1 + kn, &As[nxt][sA1]);
      }
    }

    // ---- compute current tile (8 WMMAs/wave)
    v16bf af[2];
#pragma unroll
    for (int mt = 0; mt < 2; ++mt)
      af[mt] = load_fragA(&As[cur][(wm * 32 + mt * 16 + l16) * 32], hiA);
#pragma unroll
    for (int nt = 0; nt < 4; ++nt) {
      v16bf bfr = load_fragB(&Bs[cur][(wn * 64 + nt * 16 + l16) * 32], hiB);
      acc[0][nt] = wmma_bf16(af[0], bfr, acc[0][nt]);
      acc[1][nt] = wmma_bf16(af[1], bfr, acc[1][nt]);
    }

    if (have_next) {
      if constexpr (A_F32) {
        commitA(p0, &As[nxt][sA0]);
        commitA(p1, &As[nxt][sA1]);
      }
    }
    async_fence();
    __syncthreads();
    cur = nxt;
  }

  // ---- epilogue
#pragma unroll
  for (int mt = 0; mt < 2; ++mt) {
#pragma unroll
    for (int nt = 0; nt < 4; ++nt) {
      int gn = bn + wn * 64 + nt * 16 + l16;
      float bv = bias[gn];
#pragma unroll
      for (int r = 0; r < 8; ++r) {
        int gm = bm + wm * 32 + mt * 16 + r + hiA;
        float v = acc[mt][nt][r] + bv;
        if (OUT_BF16) {
          if (gn < qcols) v *= qscale;
          ((unsigned short*)Cout)[(size_t)gm * N + gn] = f2bf(v);
        } else {
          ((float*)Cout)[(size_t)gm * N + gn] = v;
        }
      }
    }
  }
}

// ---------------------------------------------------------------------------
// Flash attention (causal).  qkv: [SEQ, 3*D_MODEL] bf16, Q pre-scaled by 1/8.
// Block: 128 threads (4 waves) -> one head, 64 query rows (16 per wave).
// Double-buffered K/V blocks; K staged async, V staged transposed via regs.
// ---------------------------------------------------------------------------
__global__ __launch_bounds__(128) void flash_attn_kernel(
    const unsigned short* __restrict__ qkv, unsigned short* __restrict__ ctx) {
  __shared__ unsigned short Ks[2][64 * 64];   // K block, row-major [key][d]
  __shared__ unsigned short Vt[2][64 * 64];   // V block, transposed [d][key]
  __shared__ unsigned short Ps[4][16 * 64];   // per-wave P bounce [m][key]

  const int ld = 3 * D_MODEL;
  const int head = blockIdx.y;
  const int qb = blockIdx.x * 64;
  const int tid = threadIdx.x;
  const int lane = tid & 31, wave = tid >> 5;
  const int l16 = lane & 15;
  const int hiA = (lane >= 16) ? 8 : 0;
  const int hiB = (lane >= 16) ? 8 : 0;

  const unsigned short* Kp = qkv + D_MODEL + head * HDIM;
  const unsigned short* Vp = qkv + 2 * D_MODEL + head * HDIM;

  // Q fragments (A-matrix, 16 rows x 64 d -> two K-steps of 32)
  v16bf qf[2];
  {
    const unsigned short* qptr =
        qkv + (size_t)(qb + wave * 16 + l16) * ld + head * HDIM;
    qf[0] = load_fragA(qptr, hiA);
    qf[1] = load_fragA(qptr + 32, hiA);
  }

  v8f o[4];
#pragma unroll
  for (int i = 0; i < 4; ++i) o[i] = (v8f){};
  float mrow[8], lrow[8];
#pragma unroll
  for (int r = 0; r < 8; ++r) { mrow[r] = -1e30f; lrow[r] = 0.f; }

  // ---- prologue: stage key-block 0 into buffer 0
#pragma unroll
  for (int j = 0; j < 4; ++j) {
    int c = tid + 128 * j;                   // 512 chunks of 8 elems
    int r = c >> 3, cc = (c & 7) * 8;
    copy16_g2l(&Kp[(size_t)r * ld + cc], &Ks[0][r * 64 + cc]);
    v8u v = *(const v8u*)&Vp[(size_t)r * ld + cc];
#pragma unroll
    for (int i = 0; i < 8; ++i) Vt[0][(cc + i) * 64 + r] = v[i];
  }
  async_fence();
  __syncthreads();

  int cur = 0;
  for (int kb = 0; kb <= qb; kb += 64) {
    const int nxt = cur ^ 1;
    const bool have_next = (kb + 64 <= qb);

    // issue next block: async K copies + V row loads into regs
    v8u vreg[4];
    if (have_next) {
#pragma unroll
      for (int j = 0; j < 4; ++j) {
        int c = tid + 128 * j;
        int r = c >> 3, cc = (c & 7) * 8;
        copy16_g2l(&Kp[(size_t)(kb + 64 + r) * ld + cc],
                   &Ks[nxt][r * 64 + cc]);
        vreg[j] = *(const v8u*)&Vp[(size_t)(kb + 64 + r) * ld + cc];
      }
    }

    // scores S[16 q, 64 k] = Q Kt  (4 key tiles x 2 K-steps of d)
    v8f s[4];
#pragma unroll
    for (int nt = 0; nt < 4; ++nt) {
      const unsigned short* kr = &Ks[cur][(nt * 16 + l16) * 64];
      v8f z = (v8f){};
      z = wmma_bf16(qf[0], load_fragB(kr, hiB), z);
      z = wmma_bf16(qf[1], load_fragB(kr + 32, hiB), z);
      s[nt] = z;
    }

    // causal mask: only the diagonal key block needs it
    if (kb == qb) {
#pragma unroll
      for (int nt = 0; nt < 4; ++nt) {
        int kg = nt * 16 + l16;
#pragma unroll
        for (int r = 0; r < 8; ++r) {
          int qg = wave * 16 + r + hiA;
          if (kg > qg) s[nt][r] = -1e30f;
        }
      }
    }

    // online softmax update (row stats per half-wave, 8 rows per lane-half)
#pragma unroll
    for (int r = 0; r < 8; ++r) {
      float mb = fmaxf(fmaxf(s[0][r], s[1][r]), fmaxf(s[2][r], s[3][r]));
      mb = hmax16(mb);
      float mn = fmaxf(mrow[r], mb);
      float rescale = __expf(mrow[r] - mn);
      mrow[r] = mn;
      float ls = 0.f;
#pragma unroll
      for (int nt = 0; nt < 4; ++nt) {
        float p = __expf(s[nt][r] - mn);
        s[nt][r] = p;
        ls += p;
      }
      ls = hsum16(ls);
      lrow[r] = lrow[r] * rescale + ls;
      o[0][r] *= rescale; o[1][r] *= rescale;
      o[2][r] *= rescale; o[3][r] *= rescale;
    }

    // bounce P through per-wave LDS: C-layout -> A-layout
    unsigned short* pw = &Ps[wave][0];
#pragma unroll
    for (int nt = 0; nt < 4; ++nt)
#pragma unroll
      for (int r = 0; r < 8; ++r)
        pw[(r + hiA) * 64 + nt * 16 + l16] = f2bf(s[nt][r]);
    asm volatile("s_wait_dscnt 0x0" ::: "memory");  // wave-local LDS RAW fence

    // O += P V   (4 d tiles x 2 K-steps of keys)
    v16bf pf0 = load_fragA(&pw[l16 * 64], hiA);
    v16bf pf1 = load_fragA(&pw[l16 * 64 + 32], hiA);
#pragma unroll
    for (int dt = 0; dt < 4; ++dt) {
      const unsigned short* vr = &Vt[cur][(dt * 16 + l16) * 64];
      o[dt] = wmma_bf16(pf0, load_fragB(vr, hiB), o[dt]);
      o[dt] = wmma_bf16(pf1, load_fragB(vr + 32, hiB), o[dt]);
    }

    // commit transposed V for next block
    if (have_next) {
#pragma unroll
      for (int j = 0; j < 4; ++j) {
        int c = tid + 128 * j;
        int r = c >> 3, cc = (c & 7) * 8;
#pragma unroll
        for (int i = 0; i < 8; ++i) Vt[nxt][(cc + i) * 64 + r] = vreg[j][i];
      }
    }
    async_fence();
    __syncthreads();
    cur = nxt;
  }

  // normalize and write ctx as bf16
#pragma unroll
  for (int r = 0; r < 8; ++r) {
    float inv = 1.0f / lrow[r];
    int qg = qb + wave * 16 + r + hiA;
#pragma unroll
    for (int dt = 0; dt < 4; ++dt)
      ctx[(size_t)qg * D_MODEL + head * HDIM + dt * 16 + l16] =
          f2bf(o[dt][r] * inv);
  }
}

// ---------------------------------------------------------------------------
// Launch
// ---------------------------------------------------------------------------
extern "C" void kernel_launch(void* const* d_in, const int* in_sizes, int n_in,
                              void* d_out, int out_size, void* d_ws, size_t ws_size,
                              hipStream_t stream) {
  (void)in_sizes; (void)n_in; (void)out_size; (void)ws_size;
  const float* x      = (const float*)d_in[0];   // [4096, 1024]
  const float* w_qkv  = (const float*)d_in[1];   // [1024, 3072]
  const float* b_qkv  = (const float*)d_in[2];   // [3072]
  const float* w_out  = (const float*)d_in[3];   // [1024, 1024]
  const float* b_out  = (const float*)d_in[4];   // [1024]
  float* out = (float*)d_out;                    // [4096, 1024]

  // workspace layout (bf16 as ushort)
  unsigned short* wqkvT = (unsigned short*)d_ws;               // [3072][1024]
  unsigned short* woutT = wqkvT + (size_t)3072 * 1024;         // [1024][1024]
  unsigned short* qkvb  = woutT + (size_t)1024 * 1024;         // [4096][3072]
  unsigned short* ctxb  = qkvb  + (size_t)4096 * 3072;         // [4096][1024]

  transpose_cvt_kernel<<<dim3(3072 / 32, 1024 / 32), dim3(32, 8), 0, stream>>>(
      w_qkv, wqkvT, 1024, 3072);
  transpose_cvt_kernel<<<dim3(1024 / 32, 1024 / 32), dim3(32, 8), 0, stream>>>(
      w_out, woutT, 1024, 1024);

  // qkv = x @ w_qkv + b_qkv, Q columns pre-scaled by 1/sqrt(64)
  gemm_bias_wmma<float, true><<<dim3(3072 / 128, 4096 / 128), 256, 0, stream>>>(
      x, wqkvT, b_qkv, (void*)qkvb, 4096, 3072, 1024, 0.125f, 1024);

  flash_attn_kernel<<<dim3(SEQ / 64, NHEAD), 128, 0, stream>>>(qkvb, ctxb);

  // out = ctx @ w_out + b_out (fp32 out)
  gemm_bias_wmma<unsigned short, false>
      <<<dim3(1024 / 128, 4096 / 128), 256, 0, stream>>>(
          ctxb, woutT, b_out, (void*)out, 4096, 1024, 1024, 1.0f, 0);
}
